// WaveFieldKV_30356828848213
// MI455X (gfx1250) — matrix operand order repaired and checked by reading, not compiled, not run on hardware
//
#include <hip/hip_runtime.h>
#include <hip/hip_bf16.h>
#include <math.h>

// Problem constants (match reference)
#define FF     64
#define SS     8192
#define DD     128
#define WW     128
#define K3     (3 * WW)          // 384
#define ALPHA  0.1f
#define LOG1MA (-0.105360515657826301f)   // log1p(-0.1)
#define TWO_PI 6.2831853071795864769f
#define INV_MAXSEQ (1.0f / 8192.0f)

typedef __attribute__((ext_vector_type(16))) __bf16 v16bf;
typedef __attribute__((ext_vector_type(8)))  float  v8f;
typedef __attribute__((ext_vector_type(4))) unsigned int u32x4;
typedef __attribute__((ext_vector_type(8))) int         i32x8;
typedef __attribute__((ext_vector_type(4))) int         i32x4;

#if __has_builtin(__builtin_amdgcn_tensor_load_to_lds) && \
    __has_builtin(__builtin_amdgcn_s_wait_tensorcnt)
#define HAS_TDM 1
#else
#define HAS_TDM 0
#endif

// ---- WMMA fragment loaders (f32 source -> bf16 fragment) -------------------
// A-matrix 16x32 bf16 layout (ISA 7.12.2): lane m = L%16 holds row M=m;
//   regs 0..3 hold K = 8*half + {0..7}, regs 4..7 hold K = 16 + 8*half + {0..7}
__device__ __forceinline__ v16bf load_frag_a(const float* __restrict__ p,
                                             int ld, int lane) {
    const int row  = lane & 15;
    const int half = lane >> 4;
    const float* q0 = p + row * ld + 8 * half;
    const float* q1 = q0 + 16;
    v16bf r;
#pragma unroll
    for (int j = 0; j < 8; ++j) r[j]     = (__bf16)q0[j];
#pragma unroll
    for (int j = 0; j < 8; ++j) r[8 + j] = (__bf16)q1[j];
    return r;
}

// B-matrix 32x16 bf16 layout: lane n = L%16 holds column N=n;
//   regs 0..3 hold K = 8*half + {0..7}, regs 4..7 hold K = 16 + 8*half + {0..7}
// Source is B^T row-major: q[n*ld + k].
__device__ __forceinline__ v16bf load_frag_bT(const float* __restrict__ p,
                                              int ld, int lane) {
    const int row  = lane & 15;
    const int half = lane >> 4;
    const float* q0 = p + row * ld + 8 * half;
    const float* q1 = q0 + 16;
    v16bf r;
#pragma unroll
    for (int j = 0; j < 8; ++j) r[j]     = (__bf16)q0[j];
#pragma unroll
    for (int j = 0; j < 8; ++j) r[8 + j] = (__bf16)q1[j];
    return r;
}

__device__ __forceinline__ v8f wmma_bf16(v16bf a, v16bf b, v8f c) {
    return __builtin_amdgcn_wmma_f32_16x16x32_bf16(
        /*neg_a=*/false, a, /*neg_b=*/false, b,
        /*c_mod=*/(short)0, c, /*reuse_a=*/false, /*reuse_b=*/false);
}

#if HAS_TDM
// Issue one TDM DMA: 16x128 f32 tile (row-contiguous) global -> LDS.
// D# per CDNA5 ISA ch.8: group0 {count, lds_addr, global_addr, type=2},
// group1 {data_size=4B, tensor 128x16, tile 128x16, dim0_stride=128}.
__device__ __forceinline__ void tdm_load_16x128(const float* gsrc,
                                                unsigned lds_off) {
    unsigned long long ga = (unsigned long long)(uintptr_t)gsrc;
    u32x4 g0;
    g0[0] = 1u;                                    // count = 1 (valid)
    g0[1] = lds_off;                               // lds_addr (bytes)
    g0[2] = (unsigned)(ga & 0xFFFFFFFFu);          // global_addr[31:0]
    g0[3] = (unsigned)((ga >> 32) & 0x01FFFFFFu)   // global_addr[56:32]
          | (2u << 30);                            // type = 2 ("image")
    i32x8 g1;
    g1[0] = (int)(2u << 16);                       // data_size code 2 = 4 bytes
    g1[1] = (int)(128u << 16);                     // tensor_dim0[15:0]  = 128
    g1[2] = (int)(16u  << 16);                     // dim0 hi=0 | tensor_dim1 lo = 16
    g1[3] = (int)(128u << 16);                     // dim1 hi=0 | tile_dim0 = 128
    g1[4] = 16;                                    // tile_dim1 = 16, tile_dim2 = 0
    g1[5] = 128;                                   // tensor_dim0_stride = 128
    g1[6] = 0;
    g1[7] = 0;
    i32x4 z4 = {0, 0, 0, 0};
#if defined(__clang_major__) && (__clang_major__ >= 23)
    i32x8 z8 = {0, 0, 0, 0, 0, 0, 0, 0};
    __builtin_amdgcn_tensor_load_to_lds(g0, g1, z4, z4, z8, 0);
#else
    __builtin_amdgcn_tensor_load_to_lds(g0, g1, z4, z4, 0);
#endif
}
#endif

// ============================================================================
// Kernel 1: fused encode GEMM + EMA-weighted reductions -> freqs/amps/phases
// grid = F workgroups, 256 threads (8 waves). Wave i owns k'-tiles
// {i (freq), 8+i (amp), 16+i (phase)} => columns w in [16i, 16i+16).
// kv is streamed via TDM into a double-buffered LDS stage (one DMA per WG
// per 16-row chunk), so all 8 waves read A fragments from LDS.
// ============================================================================
__global__ void __launch_bounds__(256)
wf_encode_reduce(const float* __restrict__ kv,        // [F,S,D]
                 const int*   __restrict__ positions, // [S]
                 const float* __restrict__ Wenc,      // [F,3W,D]
                 const float* __restrict__ basef,     // [F,W]
                 float* __restrict__ o_freq,          // [F,W]
                 float* __restrict__ o_amp,           // [F,W]
                 float* __restrict__ o_phase) {       // [F,W]
    const int f    = blockIdx.x;
    const int wave = threadIdx.x >> 5;
    const int lane = threadIdx.x & 31;
    const int half = lane >> 4;

    const float* kvf = kv   + (long)f * SS * DD;
    const float* wef = Wenc + (long)f * K3 * DD;

    // Preload B fragments for this wave's three k'-tiles, all 4 d-chunks.
    v16bf Bf[4], Ba[4], Bp[4];
#pragma unroll
    for (int dc = 0; dc < 4; ++dc) {
        Bf[dc] = load_frag_bT(wef + (( 0 + wave) * 16) * DD + dc * 32, DD, lane);
        Ba[dc] = load_frag_bT(wef + (( 8 + wave) * 16) * DD + dc * 32, DD, lane);
        Bp[dc] = load_frag_bT(wef + ((16 + wave) * 16) * DD + dc * 32, DD, lane);
    }

    // base freq for this lane's output column (C-layout N = lane%16)
    const float bf_w = basef[f * WW + wave * 16 + (lane & 15)];

    float sfreq = 0.0f, samp = 0.0f, sph = 0.0f;

#if HAS_TDM
    __shared__ float stage[2][16 * DD];     // 2 x 8 KiB double buffer
    if (wave == 0)
        tdm_load_16x128(kvf, (unsigned)(uintptr_t)&stage[0][0]);
#endif

    for (int sc = 0; sc < SS / 16; ++sc) {
        const int s0 = sc * 16;

#if HAS_TDM
        const int buf = sc & 1;
        if (wave == 0) {
            if (sc + 1 < SS / 16) {
                tdm_load_16x128(kvf + (long)(s0 + 16) * DD,
                                (unsigned)(uintptr_t)&stage[buf ^ 1][0]);
                __builtin_amdgcn_s_wait_tensorcnt((short)1);  // current buf ready
            } else {
                __builtin_amdgcn_s_wait_tensorcnt((short)0);
            }
        }
        __syncthreads();                       // publish stage[buf] to all waves
        const float* arow = &stage[buf][0];
#else
        const float* arow = kvf + (long)s0 * DD;
        if (sc + 2 < SS / 16)
            __builtin_prefetch(arow + 2 * 16 * DD + lane * 16, 0, 0);
#endif

        v8f cf, ca, cp;
#pragma unroll
        for (int i = 0; i < 8; ++i) { cf[i] = 0.0f; ca[i] = 0.0f; cp[i] = 0.0f; }

#pragma unroll
        for (int dc = 0; dc < 4; ++dc) {
            v16bf a = load_frag_a(arow + dc * 32, DD, lane);
            cf = wmma_bf16(a, Bf[dc], cf);
            ca = wmma_bf16(a, Ba[dc], ca);
            cp = wmma_bf16(a, Bp[dc], cp);
        }

        // epilogue: C-layout row M = r + 8*half, col N = lane%16
#pragma unroll
        for (int r = 0; r < 8; ++r) {
            const int   srow = s0 + r + 8 * half;
            const float wgt  = ALPHA * __expf((float)(SS - 1 - srow) * LOG1MA);
            const float pn   = (float)positions[srow] * INV_MAXSEQ;
            const float psig = __sinf(TWO_PI * bf_w * pn);
            sfreq += wgt * psig * cf[r];
            samp  += wgt * fabsf(ca[r]);
            sph   += wgt * cp[r];
        }

#if HAS_TDM
        __syncthreads();                       // stage[buf] safe to overwrite
#endif
    }

    // combine the two M-halves (lanes L and L^16 share a column)
    sfreq += __shfl_xor(sfreq, 16, 32);
    samp  += __shfl_xor(samp,  16, 32);
    sph   += __shfl_xor(sph,   16, 32);

    if (lane < 16) {
        const int w = wave * 16 + lane;
        o_freq [f * WW + w] = sfreq;
        o_amp  [f * WW + w] = samp;
        o_phase[f * WW + w] = sph;
    }
}

// ============================================================================
// Kernel 2: synthesize wave_values on the fly and decode GEMM [S,W]x[W,D].
// grid = F * (S/128) workgroups, 256 threads (8 waves); each wave owns a
// 16-row x 128-col output tile. LDS-staged coalesced b128 stores.
// ============================================================================
__global__ void __launch_bounds__(256)
wf_reconstruct_decode(const int*   __restrict__ positions, // [S]
                      const float* __restrict__ Wdec,      // [F,D,W]
                      const float* __restrict__ basef,     // [F,W]
                      const float* __restrict__ i_freq,    // [F,W]
                      const float* __restrict__ i_amp,     // [F,W]
                      const float* __restrict__ i_phase,   // [F,W]
                      float* __restrict__ out) {           // [F,S,D]
    const int SB   = SS / 128;                 // s-blocks per field
    const int f    = blockIdx.x / SB;
    const int sb   = blockIdx.x % SB;
    const int wave = threadIdx.x >> 5;
    const int lane = threadIdx.x & 31;
    const int half = lane >> 4;
    const int m    = lane & 15;                // A-layout row index
    const int s0   = sb * 128 + wave * 16;

    const float* wdf = Wdec + (long)f * DD * WW;
    const float* fr0 = i_freq  + f * WW;
    const float* am0 = i_amp   + f * WW;
    const float* ph0 = i_phase + f * WW;
    const float* bs0 = basef   + f * WW;

    const float pn = (float)positions[s0 + m] * INV_MAXSEQ;

    v8f acc[8];
#pragma unroll
    for (int nt = 0; nt < 8; ++nt)
#pragma unroll
        for (int i = 0; i < 8; ++i) acc[nt][i] = 0.0f;

#pragma unroll
    for (int kc = 0; kc < 4; ++kc) {
        // Build A fragment: a[j] = amp[k] * sin(2pi*(freq+base)[k]*pn + phase[k])
        const int kb = kc * 32 + 8 * half;
        v16bf a;
#pragma unroll
        for (int j = 0; j < 8; ++j) {
            const int k = kb + j;
            const float tf = fr0[k] + bs0[k];
            a[j] = (__bf16)(am0[k] * __sinf(TWO_PI * tf * pn + ph0[k]));
        }
#pragma unroll
        for (int j = 0; j < 8; ++j) {
            const int k = kb + 16 + j;
            const float tf = fr0[k] + bs0[k];
            a[8 + j] = (__bf16)(am0[k] * __sinf(TWO_PI * tf * pn + ph0[k]));
        }

#pragma unroll
        for (int nt = 0; nt < 8; ++nt) {
            // B^T tile: Wdec[f][nt*16 + n][kc*32 + k], k contiguous
            v16bf b = load_frag_bT(wdf + (long)(nt * 16) * WW + kc * 32, WW, lane);
            acc[nt] = wmma_bf16(a, b, acc[nt]);
        }
    }

    // Stage through LDS (8 rows per pass, stride 132 words avoids bank clash),
    // then 16B-coalesced global stores.
    __shared__ float tile[8][8 * 132];
#pragma unroll
    for (int p = 0; p < 2; ++p) {
        if (half == p) {
#pragma unroll
            for (int nt = 0; nt < 8; ++nt)
#pragma unroll
                for (int r = 0; r < 8; ++r)
                    tile[wave][r * 132 + nt * 16 + m] = acc[nt][r];
        }
        __syncthreads();
#pragma unroll
        for (int it = 0; it < 8; ++it) {
            const int row = it;            // one full 128-float row per iter
            const int g   = lane;          // 32 groups of 4 floats
            const float4 v = *(const float4*)&tile[wave][row * 132 + g * 4];
            float* dst = out + ((long)f * SS + s0 + p * 8 + row) * DD + g * 4;
            *(float4*)dst = v;
        }
        __syncthreads();
    }
}

// ============================================================================
extern "C" void kernel_launch(void* const* d_in, const int* in_sizes, int n_in,
                              void* d_out, int out_size, void* d_ws, size_t ws_size,
                              hipStream_t stream) {
    const float* kv        = (const float*)d_in[0];  // [F,S,D]
    const int*   positions = (const int*)  d_in[1];  // [S]
    const float* Wenc      = (const float*)d_in[2];  // [F,3W,D]
    const float* Wdec      = (const float*)d_in[3];  // [F,D,W]
    const float* basef     = (const float*)d_in[4];  // [F,W]
    float*       out       = (float*)d_out;          // [F,S,D]

    float* wsf    = (float*)d_ws;                    // needs 3*F*W floats (96 KiB)
    float* freqs  = wsf;
    float* amps   = wsf + FF * WW;
    float* phases = wsf + 2 * FF * WW;

    wf_encode_reduce<<<dim3(FF), dim3(256), 0, stream>>>(
        kv, positions, Wenc, basef, freqs, amps, phases);

    wf_reconstruct_decode<<<dim3(FF * (SS / 128)), dim3(256), 0, stream>>>(
        positions, Wdec, basef, freqs, amps, phases, out);
}
